// BigBirdBlock_32006096290360
// MI455X (gfx1250) — matrix operand order repaired and checked by reading, not compiled, not run on hardware
//
#include <hip/hip_runtime.h>
#include <hip/hip_bf16.h>
#include <math.h>

// ---------------------------------------------------------------------------
// BigBird transformer block for MI455X (gfx1250, wave32, WMMA bf16,
// async global->LDS staging, double-buffered GEMM pipeline).
// ---------------------------------------------------------------------------

typedef __bf16 bf16;
typedef __attribute__((ext_vector_type(8)))  bf16  bf16x8;
typedef __attribute__((ext_vector_type(16))) bf16  v16bf;
typedef __attribute__((ext_vector_type(8)))  float v8f;

#define D_MODEL 1024
#define SEQ_T   2048
#define NBATCH  2
#define NHEAD   16
#define HEAD_D  64
#define MROWS   (NBATCH * SEQ_T)   // 4096
#define DFF     4096

// ---------------- CDNA5 async global->LDS copy (ASYNCcnt-tracked) -----------

static __device__ __forceinline__ void async_ld16(unsigned lds_off,
                                                  const void* gptr) {
  unsigned long long ga = (unsigned long long)(size_t)gptr;
  asm volatile("global_load_async_to_lds_b128 %0, %1, off"
               :: "v"(lds_off), "v"(ga) : "memory");
}
static __device__ __forceinline__ void wait_async0() {
  asm volatile("s_wait_asynccnt 0" ::: "memory");
}

// ---------------- WMMA helpers (CDNA5 16x16x32 bf16, f32 accum) -------------

static __device__ __forceinline__ v16bf cat8(bf16x8 lo, bf16x8 hi) {
  return __builtin_shufflevector(lo, hi, 0,1,2,3,4,5,6,7,8,9,10,11,12,13,14,15);
}

// A-operand (16x32 bf16, ISA 7.12.2): lanes 0-15 hold K=0..7 & 16..23,
// lanes 16-31 hold K=8..15 & 24..31.  kf = (lane<16) ? 0 : 8.
static __device__ __forceinline__ v16bf ld_a(const bf16* row, int kf) {
  bf16x8 lo = *(const bf16x8*)(row + kf);
  bf16x8 hi = *(const bf16x8*)(row + kf + 16);
  return cat8(lo, hi);
}

// B-operand (32x16 bf16): per-lane column n; lanes 0-15 hold K=0..7 & 16..23,
// lanes 16-31 hold K=8..15 & 24..31 (same order as A).
static __device__ __forceinline__ v16bf ld_b(const bf16* row, int lane) {
  const bf16* p = row + ((lane & 16) ? 8 : 0);
  return cat8(*(const bf16x8*)p, *(const bf16x8*)(p + 16));
}

static __device__ __forceinline__ v8f wmma_bf(v16bf a, v16bf b, v8f c) {
  return __builtin_amdgcn_wmma_f32_16x16x32_bf16(false, a, false, b,
                                                 (short)0, c, false, false);
}

// 16-lane (half-wave) reductions: xor 1,2,4,8 stays inside each lane-group.
static __device__ __forceinline__ float rmax16(float v) {
  v = fmaxf(v, __shfl_xor(v, 1, 32));
  v = fmaxf(v, __shfl_xor(v, 2, 32));
  v = fmaxf(v, __shfl_xor(v, 4, 32));
  v = fmaxf(v, __shfl_xor(v, 8, 32));
  return v;
}
static __device__ __forceinline__ float rsum16(float v) {
  v += __shfl_xor(v, 1, 32);
  v += __shfl_xor(v, 2, 32);
  v += __shfl_xor(v, 4, 32);
  v += __shfl_xor(v, 8, 32);
  return v;
}

// ---------------- fp32 -> bf16 transposed weight conversion -----------------
// in: K x N (f32, row-major) -> out: N x K (bf16, row-major).  One 32x32 tile
// per block, LDS-staged so both global accesses are coalesced.

__global__ __launch_bounds__(256) void cvt_t_bf16(const float* __restrict__ in,
                                                  bf16* __restrict__ out,
                                                  int K, int N) {
  __shared__ float t[32][33];
  const int bx = blockIdx.x * 32;           // N base
  const int by = blockIdx.y * 32;           // K base
  const int tx = threadIdx.x & 31, ty = threadIdx.x >> 5;
  #pragma unroll
  for (int i = 0; i < 32; i += 8)
    t[ty + i][tx] = in[(size_t)(by + ty + i) * N + bx + tx];
  __syncthreads();
  #pragma unroll
  for (int i = 0; i < 32; i += 8)
    out[(size_t)(bx + ty + i) * K + by + tx] = (bf16)t[tx][ty + i];
}

// ---------------- LayerNorm (one row per block) -> bf16 ---------------------

__global__ __launch_bounds__(256) void ln_bf16(const float* __restrict__ x,
                                               const float* __restrict__ g,
                                               const float* __restrict__ bta,
                                               bf16* __restrict__ out) {
  const int row  = blockIdx.x;
  const int tid  = threadIdx.x;
  const int lane = tid & 31, w = tid >> 5;
  const float* xr = x + (size_t)row * D_MODEL;

  float s1 = 0.f, s2 = 0.f;
  #pragma unroll
  for (int i = tid; i < D_MODEL; i += 256) {
    float v = xr[i];
    s1 += v; s2 += v * v;
  }
  #pragma unroll
  for (int m = 1; m < 32; m <<= 1) {
    s1 += __shfl_xor(s1, m, 32);
    s2 += __shfl_xor(s2, m, 32);
  }
  __shared__ float a1[8], a2[8];
  if (lane == 0) { a1[w] = s1; a2[w] = s2; }
  __syncthreads();
  if (tid == 0) {
    float t1 = 0.f, t2 = 0.f;
    #pragma unroll
    for (int i = 0; i < 8; ++i) { t1 += a1[i]; t2 += a2[i]; }
    a1[0] = t1; a2[0] = t2;
  }
  __syncthreads();
  const float mu  = a1[0] * (1.f / D_MODEL);
  const float var = a2[0] * (1.f / D_MODEL) - mu * mu;
  const float rs  = rsqrtf(var + 1e-5f);
  bf16* orow = out + (size_t)row * D_MODEL;
  #pragma unroll
  for (int i = tid; i < D_MODEL; i += 256)
    orow[i] = (bf16)((xr[i] - mu) * rs * g[i] + bta[i]);
}

// ---------------- bf16 WMMA GEMM with async double-buffered staging ---------
// C(MxN) = A(MxK,bf16) * Bt(NxK,bf16 = B^T)  [+bias][GELU][+resid] -> f32/bf16
// Block tile 128x128x32, 8 waves (2x4), wave tile 64x32.  Both tiles staged
// by GLOBAL_LOAD_ASYNC_TO_LDS_B128 into the back buffer while WMMAs consume
// the front buffer.

#define BM 128
#define BN 128
#define BK 32
#define LDA 40   // padded K-stride (bf16 elems): 80B rows, 16B aligned
#define LDB 40

__global__ __launch_bounds__(256) void gemm_bf16(
    const bf16* __restrict__ A, const bf16* __restrict__ Bt,
    const float* __restrict__ bias, const float* __restrict__ resid,
    float* __restrict__ outF, bf16* __restrict__ outB,
    int M, int N, int K, int act_gelu) {
  __shared__ bf16 As[2][BM * LDA];    // row-major MxK tiles (double buffered)
  __shared__ bf16 Bs[2][BN * LDB];    // row-major NxK tiles (B^T)

  const int tid  = threadIdx.x;
  const int lane = tid & 31, wave = tid >> 5;
  const int wr   = wave >> 2, wc = wave & 3;       // 2 x 4 wave grid
  const int bm   = blockIdx.y * BM, bn = blockIdx.x * BN;
  const int l16  = lane & 15, kf = (lane & 16) ? 8 : 0;

  v8f acc[4][2];
  #pragma unroll
  for (int i = 0; i < 4; ++i)
    #pragma unroll
    for (int j = 0; j < 2; ++j) acc[i][j] = {};

  // 1024 16B chunks per tile pair per buffer: 2 A-chunks + 2 B-chunks / thread.
#define STAGE(buf, kk)                                                         \
  do {                                                                         \
    _Pragma("unroll")                                                          \
    for (int i_ = 0; i_ < 2; ++i_) {                                           \
      int idx_ = tid * 2 + i_;                                                 \
      int r_ = idx_ >> 2, kc_ = (idx_ & 3) * 8;                                \
      async_ld16((unsigned)(size_t)&As[buf][r_ * LDA + kc_],                   \
                 A + (size_t)(bm + r_) * K + (kk) + kc_);                      \
      async_ld16((unsigned)(size_t)&Bs[buf][r_ * LDB + kc_],                   \
                 Bt + (size_t)(bn + r_) * K + (kk) + kc_);                     \
    }                                                                          \
  } while (0)

  STAGE(0, 0);
  wait_async0();
  __syncthreads();

  int cur = 0;
  for (int k0 = 0; k0 < K; k0 += BK) {
    v16bf aop[4], bop[2];
    #pragma unroll
    for (int mt = 0; mt < 4; ++mt)
      aop[mt] = ld_a(&As[cur][(wr * 64 + mt * 16 + l16) * LDA], kf);
    #pragma unroll
    for (int nt = 0; nt < 2; ++nt)
      bop[nt] = ld_b(&Bs[cur][(wc * 32 + nt * 16 + l16) * LDB], lane);

    if (k0 + BK < K) STAGE(cur ^ 1, k0 + BK);   // prefetch next tile

    #pragma unroll
    for (int mt = 0; mt < 4; ++mt)
      #pragma unroll
      for (int nt = 0; nt < 2; ++nt)
        acc[mt][nt] = wmma_bf(aop[mt], bop[nt], acc[mt][nt]);

    wait_async0();
    __syncthreads();
    cur ^= 1;
  }
#undef STAGE

  // Epilogue.  D layout: VGPR r -> row (r + 8*(lane>=16)), lane%16 -> col.
  const int grp = (lane & 16) ? 8 : 0;
  const bool hasBias = bias != nullptr, hasRes = resid != nullptr;
  #pragma unroll
  for (int mt = 0; mt < 4; ++mt) {
    #pragma unroll
    for (int nt = 0; nt < 2; ++nt) {
      #pragma unroll
      for (int r = 0; r < 8; ++r) {
        int row = bm + wr * 64 + mt * 16 + r + grp;
        int col = bn + wc * 32 + nt * 16 + l16;
        float v = acc[mt][nt][r];
        if (hasBias) v += bias[col];
        if (act_gelu) v = 0.5f * v * (1.f + erff(v * 0.70710678118654752f));
        if (hasRes)  v += resid[(size_t)row * N + col];
        size_t o = (size_t)row * N + col;
        if (outF) outF[o] = v;
        if (outB) outB[o] = (bf16)v;
      }
    }
  }
}

// ---------------- BigBird masked flash attention ----------------------------
// Grid: (T/64, B*H).  128 threads = 4 waves; wave w owns 16 queries.
// Key tiles of 32 (= one WMMA K).  All-masked 64q x 32k tiles are skipped
// block-uniformly (BigBird mask is ~2% dense).  K tile staged via async.

__global__ __launch_bounds__(128) void bigbird_attn(
    const bf16* __restrict__ q, const bf16* __restrict__ k,
    const bf16* __restrict__ v, const unsigned char* __restrict__ mask,
    bf16* __restrict__ out) {
  __shared__ bf16 kt[32 * 80];        // 32 keys x 64 hd  (stride 80: 16B rows)
  __shared__ bf16 vt[64 * 40];        // V^T: 64 hd x 32 keys (stride 40)
  __shared__ bf16 pt[4][16 * 40];     // per-wave P tile 16x32 (stride 40)
  __shared__ int  sAny;

  const int tid  = threadIdx.x;
  const int lane = tid & 31, w = tid >> 5;
  const int l16  = lane & 15;
  const int kf   = (lane & 16) ? 8 : 0;
  const int grp  = (lane & 16) ? 8 : 0;

  const int qb = blockIdx.x * 64;
  const int b  = blockIdx.y >> 4, h = blockIdx.y & 15;
  const size_t rowbase = (size_t)b * SEQ_T;
  const int col0 = h * HEAD_D;

  // Q tile (16x64) for this wave, straight from global in A-operand layout.
  const int qrow = qb + w * 16 + l16;
  const bf16* qp = q + (rowbase + qrow) * D_MODEL + col0;
  const v16bf aq0 = ld_a(qp, kf);        // hd 0..31
  const v16bf aq1 = ld_a(qp + 32, kf);   // hd 32..63

  float m8[8], l8[8];
  v8f o[4];
  #pragma unroll
  for (int r = 0; r < 8; ++r) { m8[r] = -3.0e38f; l8[r] = 0.f; }
  #pragma unroll
  for (int j = 0; j < 4; ++j) o[j] = {};

  for (int k0 = 0; k0 < SEQ_T; k0 += 32) {
    // Per-lane mask bytes: rows (qb + w*16 + grp + r), cols k0+l16 / +16.
    // The 128 threads exactly partition the 64x32 mask tile.
    unsigned char mb[16];
    int any = 0;
    #pragma unroll
    for (int r = 0; r < 8; ++r) {
      const unsigned char* mp =
          mask + (size_t)(qb + w * 16 + grp + r) * SEQ_T + k0 + l16;
      mb[r]     = mp[0];
      mb[8 + r] = mp[16];
      any |= mb[r] | mb[8 + r];
    }
    __syncthreads();                     // prev compute done (kt/vt reusable)
    if (tid == 0) sAny = 0;
    __syncthreads();
    if (any) sAny = 1;
    __syncthreads();
    if (!sAny) continue;                 // block-uniform skip of empty tile

    // Stage K tile (32 keys x 64 hd) row-major via async global->LDS.
    #pragma unroll
    for (int i = 0; i < 2; ++i) {
      int idx = tid * 2 + i;             // 0..255
      int key = idx >> 3, hd = (idx & 7) * 8;
      async_ld16((unsigned)(size_t)&kt[key * 80 + hd],
                 k + (rowbase + k0 + key) * D_MODEL + col0 + hd);
    }
    // Stage V^T (64 hd x 32 keys) with a register transpose.
    {
      int key = tid & 31, hb = (tid >> 5) * 16;
      const bf16* vp = v + (rowbase + k0 + key) * D_MODEL + col0 + hb;
      bf16x8 v0 = *(const bf16x8*)(vp);
      bf16x8 v1 = *(const bf16x8*)(vp + 8);
      #pragma unroll
      for (int j = 0; j < 8; ++j) {
        vt[(hb + j) * 40 + key]     = v0[j];
        vt[(hb + 8 + j) * 40 + key] = v1[j];
      }
    }
    wait_async0();
    __syncthreads();

    // Scores: S(16q x 32k) = Q(16x64) K^T, as two 16x16 D tiles.
    v16bf b0 = ld_b(kt + l16 * 80,             lane);   // keys 0-15, hd 0-31
    v16bf b1 = ld_b(kt + l16 * 80 + 32,        lane);   // keys 0-15, hd 32-63
    v16bf b2 = ld_b(kt + (16 + l16) * 80,      lane);   // keys 16-31
    v16bf b3 = ld_b(kt + (16 + l16) * 80 + 32, lane);
    v8f z = {};
    v8f s_lo = wmma_bf(aq1, b1, wmma_bf(aq0, b0, z));
    v8f s_hi = wmma_bf(aq1, b3, wmma_bf(aq0, b2, z));
    s_lo *= 0.125f;                      // 1/sqrt(64)
    s_hi *= 0.125f;

    // Mask + online softmax (rows live in VGPR index r, group in lane/16).
    #pragma unroll
    for (int r = 0; r < 8; ++r) {
      if (!mb[r])     s_lo[r] = -1.0e30f;
      if (!mb[8 + r]) s_hi[r] = -1.0e30f;
      float tmax = rmax16(fmaxf(s_lo[r], s_hi[r]));
      float nm   = fmaxf(m8[r], tmax);
      float al   = __expf(m8[r] - nm);
      float plo  = __expf(s_lo[r] - nm);
      float phi  = __expf(s_hi[r] - nm);
      l8[r] = l8[r] * al + rsum16(plo + phi);
      m8[r] = nm;
      o[0][r] *= al; o[1][r] *= al; o[2][r] *= al; o[3][r] *= al;
      pt[w][(r + grp) * 40 + l16]      = (bf16)plo;
      pt[w][(r + grp) * 40 + 16 + l16] = (bf16)phi;
    }

    // O += P(16x32) * V(32x64): one A operand, four B operands.
    v16bf pa = ld_a(pt[w] + l16 * 40, kf);
    #pragma unroll
    for (int j = 0; j < 4; ++j) {
      v16bf bv = ld_b(vt + (j * 16 + l16) * 40, lane);
      o[j] = wmma_bf(pa, bv, o[j]);
    }
  }

  // Normalize and write bf16 attention output (row = token, col = h*64+hd).
  #pragma unroll
  for (int r = 0; r < 8; ++r) {
    float inv = 1.f / l8[r];
    int row = qb + w * 16 + grp + r;
    #pragma unroll
    for (int j = 0; j < 4; ++j)
      out[(rowbase + row) * D_MODEL + col0 + j * 16 + l16] = (bf16)(o[j][r] * inv);
  }
}

// ---------------- Host-side orchestration -----------------------------------

extern "C" void kernel_launch(void* const* d_in, const int* in_sizes, int n_in,
                              void* d_out, int out_size, void* d_ws, size_t ws_size,
                              hipStream_t stream) {
  const float*         x    = (const float*)d_in[0];
  const unsigned char* mask = (const unsigned char*)d_in[1];
  const float* ln1_g = (const float*)d_in[2];
  const float* ln1_b = (const float*)d_in[3];
  const float* ln2_g = (const float*)d_in[4];
  const float* ln2_b = (const float*)d_in[5];
  const float* Wq = (const float*)d_in[6];
  const float* Wk = (const float*)d_in[7];
  const float* Wv = (const float*)d_in[8];
  const float* Wo = (const float*)d_in[9];
  const float* bo = (const float*)d_in[10];
  const float* W1 = (const float*)d_in[11];
  const float* b1 = (const float*)d_in[12];
  const float* W2 = (const float*)d_in[13];
  const float* b2 = (const float*)d_in[14];
  float* outp = (float*)d_out;

  // Workspace carve-up (256B aligned).
  char* ws = (char*)d_ws;
  size_t off = 0;
  auto carve = [&](size_t bytes) -> char* {
    char* p = ws + off;
    off += (bytes + 255) & ~(size_t)255;
    return p;
  };
  const size_t DD = (size_t)D_MODEL * D_MODEL;           // 1M
  const size_t DF = (size_t)D_MODEL * DFF;               // 4M
  const size_t MD = (size_t)MROWS * D_MODEL;             // 4M
  const size_t MF = (size_t)MROWS * DFF;                 // 16M
  bf16*  wq_t = (bf16*)carve(DD * 2);   // all weights stored transposed (NxK)
  bf16*  wk_t = (bf16*)carve(DD * 2);
  bf16*  wv_t = (bf16*)carve(DD * 2);
  bf16*  wo_t = (bf16*)carve(DD * 2);
  bf16*  w1_t = (bf16*)carve(DF * 2);
  bf16*  w2_t = (bf16*)carve(DF * 2);
  bf16*  h_b  = (bf16*)carve(MD * 2);
  bf16*  q_b  = (bf16*)carve(MD * 2);
  bf16*  k_b  = (bf16*)carve(MD * 2);
  bf16*  v_b  = (bf16*)carve(MD * 2);
  bf16*  ao_b = (bf16*)carve(MD * 2);
  float* x2   = (float*)carve(MD * 4);
  bf16*  h2_b = (bf16*)carve(MD * 2);
  bf16*  f1_b = (bf16*)carve(MF * 2);
  (void)ws_size; (void)in_sizes; (void)n_in; (void)out_size;

  // 1) Weights -> transposed bf16 (N x K).
  dim3 c256(256);
  cvt_t_bf16<<<dim3(D_MODEL/32, D_MODEL/32), c256, 0, stream>>>(Wq, wq_t, D_MODEL, D_MODEL);
  cvt_t_bf16<<<dim3(D_MODEL/32, D_MODEL/32), c256, 0, stream>>>(Wk, wk_t, D_MODEL, D_MODEL);
  cvt_t_bf16<<<dim3(D_MODEL/32, D_MODEL/32), c256, 0, stream>>>(Wv, wv_t, D_MODEL, D_MODEL);
  cvt_t_bf16<<<dim3(D_MODEL/32, D_MODEL/32), c256, 0, stream>>>(Wo, wo_t, D_MODEL, D_MODEL);
  cvt_t_bf16<<<dim3(DFF/32,     D_MODEL/32), c256, 0, stream>>>(W1, w1_t, D_MODEL, DFF);
  cvt_t_bf16<<<dim3(D_MODEL/32, DFF/32),     c256, 0, stream>>>(W2, w2_t, DFF, D_MODEL);

  // 2) h = LN1(x) as bf16.
  ln_bf16<<<MROWS, 256, 0, stream>>>(x, ln1_g, ln1_b, h_b);

  // 3) Q/K/V projections.
  dim3 gQ(D_MODEL / BN, MROWS / BM);
  gemm_bf16<<<gQ, 256, 0, stream>>>(h_b, wq_t, nullptr, nullptr, nullptr, q_b,
                                    MROWS, D_MODEL, D_MODEL, 0);
  gemm_bf16<<<gQ, 256, 0, stream>>>(h_b, wk_t, nullptr, nullptr, nullptr, k_b,
                                    MROWS, D_MODEL, D_MODEL, 0);
  gemm_bf16<<<gQ, 256, 0, stream>>>(h_b, wv_t, nullptr, nullptr, nullptr, v_b,
                                    MROWS, D_MODEL, D_MODEL, 0);

  // 4) Masked attention.
  dim3 gA(SEQ_T / 64, NBATCH * NHEAD);
  bigbird_attn<<<gA, 128, 0, stream>>>(q_b, k_b, v_b, mask, ao_b);

  // 5) x2 = x + attn_out @ Wo + bo   (f32).
  gemm_bf16<<<gQ, 256, 0, stream>>>(ao_b, wo_t, bo, x, x2, nullptr,
                                    MROWS, D_MODEL, D_MODEL, 0);

  // 6) h2 = LN2(x2) as bf16.
  ln_bf16<<<MROWS, 256, 0, stream>>>(x2, ln2_g, ln2_b, h2_b);

  // 7) f1 = GELU(h2 @ W1 + b1) as bf16.
  dim3 gF1(DFF / BN, MROWS / BM);
  gemm_bf16<<<gF1, 256, 0, stream>>>(h2_b, w1_t, b1, nullptr, nullptr, f1_b,
                                     MROWS, DFF, D_MODEL, 1);

  // 8) out = x2 + f1 @ W2 + b2   (f32).
  gemm_bf16<<<gQ, 256, 0, stream>>>(f1_b, w2_t, b2, x2, outp, nullptr,
                                    MROWS, D_MODEL, DFF, 0);
}
